// Event_Model_88923002896870
// MI455X (gfx1250) — matrix operand order repaired and checked by reading, not compiled, not run on hardware
//
#include <hip/hip_runtime.h>
#include <hip/hip_bf16.h>

typedef __attribute__((ext_vector_type(16))) __bf16 v16bf;
typedef __attribute__((ext_vector_type(8)))  float  v8f;
typedef unsigned int v4u __attribute__((ext_vector_type(4)));
typedef int          v8i __attribute__((ext_vector_type(8)));
typedef int          v4i __attribute__((ext_vector_type(4)));

union BF16Frag {
    v16bf v;
    uint4 q[2];
    unsigned short s[16];
};

__device__ __forceinline__ float bf2f(unsigned short h) {
    return __uint_as_float(((unsigned int)h) << 16);
}
__device__ __forceinline__ unsigned short f2bf(float f) {
    unsigned int u = __float_as_uint(f);
    u += 0x7FFFu + ((u >> 16) & 1u);   // round-to-nearest-even
    return (unsigned short)(u >> 16);
}

// ---------------------------------------------------------------------------
// Tensor Data Mover: 2-D tile (rows x cols of bf16) global -> LDS.
// D# per CDNA5 ISA:
//   group0: count=1 | lds_addr[63:32] | global_addr[120:64] | type=2[127:126]
//   group1: data_size=1(2B)[17:16] | tensor_dim0[79:48] | tensor_dim1[111:80]
//           | tile_dim0[127:112] | tile_dim1[143:128] | dim0_stride[207:160]
// This toolchain uses the 6-arg builtin (g0, g1, g2, g3, g_extra, cpol).
// ---------------------------------------------------------------------------
__device__ __forceinline__ void tdm_load_2d_bf16(unsigned lds_off, const unsigned short* gptr,
                                                 unsigned tensor_d0, unsigned tensor_d1,
                                                 unsigned tile_d0, unsigned tile_d1,
                                                 unsigned stride0) {
    unsigned long long ga = (unsigned long long)(size_t)gptr;
    v4u g0;
    g0.x = 1u;                                   // count=1 (valid), user mode
    g0.y = lds_off;                              // LDS byte address
    g0.z = (unsigned)(ga & 0xFFFFFFFFu);         // global_addr[31:0]
    g0.w = (unsigned)(ga >> 32) | (2u << 30);    // global_addr[56:32] | type=2
    v8i g1;
    g1[0] = (int)(1u << 16);                     // data_size = 1 -> 2 bytes
    g1[1] = (int)((tensor_d0 & 0xFFFFu) << 16);  // tensor_dim0[15:0] @ bits 63:48
    g1[2] = (int)((tensor_d0 >> 16) | ((tensor_d1 & 0xFFFFu) << 16));
    g1[3] = (int)((tensor_d1 >> 16) | (tile_d0 << 16));   // tile_dim0 @ 127:112
    g1[4] = (int)tile_d1;                        // tile_dim1 @ 143:128 (tile_dim2=0)
    g1[5] = (int)stride0;                        // tensor_dim0_stride[31:0]
    g1[6] = 0;                                   // stride0 hi | stride1 lo
    g1[7] = 0;
    v4i z4 = {0, 0, 0, 0};
    v8i z8 = {0, 0, 0, 0, 0, 0, 0, 0};
    __builtin_amdgcn_tensor_load_to_lds(g0, g1, z4, z4, z8, 0);
}

// ---------------------------------------------------------------------------
// f32 -> bf16 (row-major copy, for GEMM A operands)
// ---------------------------------------------------------------------------
__global__ void k_cvt_bf16(const float* __restrict__ src, unsigned short* __restrict__ dst, int n) {
    int i = blockIdx.x * 256 + threadIdx.x;
    if (i < n) dst[i] = f2bf(src[i]);
}

// f32 [K,Nsrc] -> bf16 transposed [N,K]; rows n >= Nsrc zero-padded
__global__ void k_cvt_T(const float* __restrict__ src, unsigned short* __restrict__ dst,
                        int K, int N, int Nsrc) {
    int i = blockIdx.x * 256 + threadIdx.x;
    if (i >= N * K) return;
    int n = i / K, k = i - n * K;
    dst[i] = (n < Nsrc) ? f2bf(src[(size_t)k * Nsrc + n]) : (unsigned short)0;
}

__global__ void k_pad_bias(const float* __restrict__ b, float* __restrict__ bp) {
    int i = blockIdx.x * 256 + threadIdx.x;
    if (i < 512) bp[i] = (i < 500) ? b[i] : 0.f;
}

// ---------------------------------------------------------------------------
// bf16 WMMA GEMM: C = relu?(A[M,K] @ B[K,N] + bias), B given transposed BT[N,K].
// WG tile 128x64, 8 waves (4x2), each wave 32x32 = 4 wmma accumulators.
// A tile (128x32) staged by the Tensor Data Mover (wave 0 issues descriptor);
// B tile staged as vectorized b128 LDS copies from BT. Next B tile prefetched.
// ---------------------------------------------------------------------------
__global__ __launch_bounds__(256)
void k_gemm_bf16(const unsigned short* __restrict__ A, const unsigned short* __restrict__ BT,
                 const float* __restrict__ bias, int M, int N, int K,
                 unsigned short* __restrict__ Cb, float* __restrict__ Cf,
                 int relu, int rowMul, int rowAdd, int ldC) {
    __shared__ __attribute__((aligned(16))) unsigned short As[128 * 32];
    __shared__ __attribute__((aligned(16))) unsigned short Bs[64 * 32];   // Bs[col*32 + k]

    const int tid  = threadIdx.x;
    const int lane = tid & 31;
    const int wave = tid >> 5;
    const int m0 = blockIdx.y * 128;
    const int n0 = blockIdx.x * 64;
    const int wm = (wave >> 1) * 32;
    const int wn = (wave & 1) * 32;
    const int lrow  = lane & 15;
    const int kb16  = (lane < 16) ? 0 : 8;     // A-frag K base (16-bit A 16x32 layout)
    const int kb32  = (lane < 16) ? 0 : 16;    // unused; B frag uses kb16, same k order as A
    const int rbase = (lane < 16) ? 0 : 8;     // C/D row base

    const int bcol = tid >> 2;                  // B staging: 8 halfs (16B) per thread
    const int bq   = (tid & 3) * 8;
    const unsigned asOff = (unsigned)(size_t)&As[0];

    v8f acc[2][2] = {};

    for (int kc = 0; kc < K; kc += 32) {
        // A tile via TDM: rows m0..m0+127, cols kc..kc+31 of A[M,K]
        if (wave == 0)
            tdm_load_2d_bf16(asOff, A + (size_t)m0 * K + kc,
                             (unsigned)K, (unsigned)M, 32u, 128u, (unsigned)K);
        // B tile: BT[n0+bcol][kc+bq .. +7] -> Bs[bcol*32 + bq]
        const unsigned short* bsrc = BT + (size_t)(n0 + bcol) * K + kc + bq;
        *(uint4*)(Bs + bcol * 32 + bq) = *(const uint4*)bsrc;
        if (kc + 32 < K) __builtin_prefetch(bsrc + 32, 0, 1);   // global_prefetch_b8
        __builtin_amdgcn_s_wait_tensorcnt(0);
        __syncthreads();

        BF16Frag af[2], bf[2];
#pragma unroll
        for (int mt = 0; mt < 2; ++mt) {
            const unsigned short* ap = As + (wm + mt * 16 + lrow) * 32 + kb16;
            af[mt].q[0] = *(const uint4*)ap;
            af[mt].q[1] = *(const uint4*)(ap + 16);
        }
#pragma unroll
        for (int nt = 0; nt < 2; ++nt) {
            const unsigned short* bp = Bs + (wn + nt * 16 + lrow) * 32 + kb16;
            bf[nt].q[0] = *(const uint4*)bp;
            bf[nt].q[1] = *(const uint4*)(bp + 16);
        }
#pragma unroll
        for (int mt = 0; mt < 2; ++mt)
#pragma unroll
            for (int nt = 0; nt < 2; ++nt)
                acc[mt][nt] = __builtin_amdgcn_wmma_f32_16x16x32_bf16(
                    false, af[mt].v, false, bf[nt].v, (short)0, acc[mt][nt], false, false);
        __syncthreads();
    }

#pragma unroll
    for (int mt = 0; mt < 2; ++mt)
#pragma unroll
        for (int nt = 0; nt < 2; ++nt) {
            const int col = n0 + wn + nt * 16 + lrow;
            const float bv = bias[col];
#pragma unroll
            for (int r = 0; r < 8; ++r) {
                int row = m0 + wm + mt * 16 + rbase + r;
                float v = acc[mt][nt][r] + bv;
                if (relu) v = fmaxf(v, 0.f);
                size_t orow = (size_t)row * rowMul + rowAdd;
                if (Cf) Cf[orow * ldC + col] = v;
                else    Cb[orow * ldC + col] = f2bf(v);
            }
        }
}

// ---------------------------------------------------------------------------
// Per-graph GCN: one workgroup (256 thr, 8 waves) per graph of 16 nodes x 512.
// gc weights are pre-transposed: gc1T/gc3T [32][512], gc2T/gc4T [512][32].
// x tile loaded by TDM. Produces relu(feat) bf16 (fused D-mean + fusion).
// ---------------------------------------------------------------------------
__global__ __launch_bounds__(256)
void k_graph(const unsigned short* __restrict__ objb,
             const unsigned short* __restrict__ gc1T, const float* __restrict__ gc1b,
             const unsigned short* __restrict__ gc2T, const float* __restrict__ gc2b,
             const unsigned short* __restrict__ gc3T, const float* __restrict__ gc3b,
             const unsigned short* __restrict__ gc4T, const float* __restrict__ gc4b,
             unsigned short* __restrict__ featb) {
    __shared__ __attribute__((aligned(16))) unsigned short xb[16 * 512];
    __shared__ float tbuf[16 * 512];
    __shared__ float res[16 * 512];
    __shared__ float adjA[256];
    __shared__ float adjB[256];
    __shared__ float h1f[16 * 32];
    __shared__ float h3f[16 * 32];
    __shared__ float y1f[16 * 32];
    __shared__ float y3f[16 * 32];
    __shared__ __attribute__((aligned(16))) unsigned short y1b[16 * 32];
    __shared__ __attribute__((aligned(16))) unsigned short y3b[16 * 32];

    const int tid  = threadIdx.x;
    const int lane = tid & 31;
    const int wave = tid >> 5;
    const int g    = blockIdx.x;
    const int lrow  = lane & 15;
    const int kb16  = (lane < 16) ? 0 : 8;
    const int kb32  = (lane < 16) ? 0 : 16;
    const int rbase = (lane < 16) ? 0 : 8;
    const int ai = tid >> 4, aj = tid & 15;

    // 1) x tile (16x512 bf16 = 16KB, contiguous) via TDM
    if (wave == 0)
        tdm_load_2d_bf16((unsigned)(size_t)&xb[0], objb + (size_t)g * 16 * 512,
                         8192u, 1u, 8192u, 1u, 8192u);
    __builtin_amdgcn_s_wait_tensorcnt(0);
    __syncthreads();

    // 2) s = x @ x^T (one element per thread)
    {
        float s = 0.f;
        for (int k = 0; k < 512; ++k)
            s += bf2f(xb[ai * 512 + k]) * bf2f(xb[aj * 512 + k]);
        adjA[tid] = s;
    }
    __syncthreads();
    // 3) exp(s - rowmax)
    {
        float m = -1e30f;
        for (int k = 0; k < 16; ++k) m = fmaxf(m, adjA[ai * 16 + k]);
        adjB[tid] = __expf(adjA[tid] - m);
    }
    __syncthreads();
    // 4) D^{-1/2} A D^{-1/2}
    {
        float di = 0.f, dj = 0.f;
        for (int k = 0; k < 16; ++k) { di += adjB[ai * 16 + k]; dj += adjB[aj * 16 + k]; }
        adjA[tid] = adjB[tid] * rsqrtf(di) * rsqrtf(dj);
    }
    __syncthreads();

    // 5) h1 = x @ gc1_W, h3 = x @ gc3_W (WMMA; waves 0..3, K=512 in 16 steps)
    if (wave < 4) {
        const int br = wave >> 1;
        const int nt = wave & 1;
        const unsigned short* W = br ? gc3T : gc1T;   // [32 cols][512]
        float* H = br ? h3f : h1f;
        const int col = nt * 16 + lrow;
        v8f acc = {};
        for (int kc = 0; kc < 16; ++kc) {
            BF16Frag af;
            const unsigned short* ap = xb + lrow * 512 + kc * 32 + kb16;
            af.q[0] = *(const uint4*)ap;
            af.q[1] = *(const uint4*)(ap + 16);
            BF16Frag bf;
            const unsigned short* bp = W + (size_t)col * 512 + kc * 32 + kb16;
            bf.q[0] = *(const uint4*)bp;
            bf.q[1] = *(const uint4*)(bp + 16);
            acc = __builtin_amdgcn_wmma_f32_16x16x32_bf16(false, af.v, false, bf.v, (short)0, acc, false, false);
        }
#pragma unroll
        for (int r = 0; r < 8; ++r) H[(rbase + r) * 32 + col] = acc[r];
    }
    __syncthreads();

    // 6) y = relu(adj @ h + b), keep f32 + bf16 copies
#pragma unroll
    for (int q = 0; q < 2; ++q) {
        int idx = tid + q * 256;
        int i = idx >> 5, n = idx & 31;
        float s1 = gc1b[n], s3 = gc3b[n];
        for (int j = 0; j < 16; ++j) {
            float a = adjA[i * 16 + j];
            s1 += a * h1f[j * 32 + n];
            s3 += a * h3f[j * 32 + n];
        }
        s1 = fmaxf(s1, 0.f); s3 = fmaxf(s3, 0.f);
        y1f[idx] = s1; y3f[idx] = s3;
        y1b[idx] = f2bf(s1); y3b[idx] = f2bf(s3);
    }
    __syncthreads();

    // 7) t1 = y1 @ gc2_W (16x32 @ 32x512, 32 N-tiles, 4 per wave)
    {
        BF16Frag af;
        const unsigned short* ap = y1b + lrow * 32 + kb16;
        af.q[0] = *(const uint4*)ap;
        af.q[1] = *(const uint4*)(ap + 16);
#pragma unroll
        for (int q = 0; q < 4; ++q) {
            const int col = (wave * 4 + q) * 16 + lrow;
            BF16Frag bf;
            const unsigned short* bp = gc2T + (size_t)col * 32 + kb16;   // [512 cols][32]
            bf.q[0] = *(const uint4*)bp;
            bf.q[1] = *(const uint4*)(bp + 16);
            v8f acc = {};
            acc = __builtin_amdgcn_wmma_f32_16x16x32_bf16(false, af.v, false, bf.v, (short)0, acc, false, false);
#pragma unroll
            for (int r = 0; r < 8; ++r) tbuf[(rbase + r) * 512 + col] = acc[r];
        }
    }
    __syncthreads();

    // 8) res = adj @ t1 + gc2_b
    {
        const int i = tid >> 4;
        const int cb = (tid & 15) * 32;
        for (int cc = 0; cc < 32; ++cc) {
            int c = cb + cc;
            float s = gc2b[c];
            for (int j = 0; j < 16; ++j) s += adjA[i * 16 + j] * tbuf[j * 512 + c];
            res[i * 512 + c] = s;
        }
    }
    __syncthreads();

    // 9) adj_hat3 from y3: y22, row norms, 1 + y22/(n n^T), D^{-1/2} norm
    {
        float v = 0.f;
        for (int k = 0; k < 32; ++k) v += y3f[ai * 32 + k] * y3f[aj * 32 + k];
        adjB[tid] = v;
    }
    __syncthreads();
    {
        float ni = 0.f, nj = 0.f;
        for (int k = 0; k < 16; ++k) {
            float a = adjB[ai * 16 + k]; ni += a * a;
            float b = adjB[aj * 16 + k]; nj += b * b;
        }
        adjA[tid] = 1.f + adjB[tid] / (sqrtf(ni) * sqrtf(nj));
    }
    __syncthreads();
    {
        float di = 0.f, dj = 0.f;
        for (int k = 0; k < 16; ++k) { di += adjA[ai * 16 + k]; dj += adjA[aj * 16 + k]; }
        adjB[tid] = adjA[tid] * rsqrtf(di) * rsqrtf(dj);
    }
    __syncthreads();

    // 10) t2 = y3 @ gc4_W
    {
        BF16Frag af;
        const unsigned short* ap = y3b + lrow * 32 + kb16;
        af.q[0] = *(const uint4*)ap;
        af.q[1] = *(const uint4*)(ap + 16);
#pragma unroll
        for (int q = 0; q < 4; ++q) {
            const int col = (wave * 4 + q) * 16 + lrow;
            BF16Frag bf;
            const unsigned short* bp = gc4T + (size_t)col * 32 + kb16;
            bf.q[0] = *(const uint4*)bp;
            bf.q[1] = *(const uint4*)(bp + 16);
            v8f acc = {};
            acc = __builtin_amdgcn_wmma_f32_16x16x32_bf16(false, af.v, false, bf.v, (short)0, acc, false, false);
#pragma unroll
            for (int r = 0; r < 8; ++r) tbuf[(rbase + r) * 512 + col] = acc[r];
        }
    }
    __syncthreads();

    // 11) res += adj_hat3 @ t2 + gc4_b
    {
        const int i = tid >> 4;
        const int cb = (tid & 15) * 32;
        for (int cc = 0; cc < 32; ++cc) {
            int c = cb + cc;
            float s = gc4b[c];
            for (int j = 0; j < 16; ++j) s += adjB[i * 16 + j] * tbuf[j * 512 + c];
            res[i * 512 + c] += s;
        }
    }
    __syncthreads();

    // 12) feat = relu(0.5*(mean_D(obj) + mean_D(0.5*(x1+y2)))), store bf16
#pragma unroll
    for (int q = 0; q < 2; ++q) {
        int c = tid + q * 256;
        float o = 0.f, gr = 0.f;
        for (int i = 0; i < 16; ++i) {
            o  += bf2f(xb[i * 512 + c]);
            gr += res[i * 512 + c];
        }
        o *= (1.f / 16.f);
        gr *= 0.5f * (1.f / 16.f);
        float f = fmaxf(0.5f * (o + gr), 0.f);
        featb[(size_t)g * 512 + c] = f2bf(f);
    }
}

// ---------------------------------------------------------------------------
// out[n][k] = mean over T of logits[(n*64+t)*512 + k], k < 500
// ---------------------------------------------------------------------------
__global__ void k_reduce_t(const float* __restrict__ logits, float* __restrict__ out) {
    int idx = blockIdx.x * 256 + threadIdx.x;
    if (idx >= 32 * 500) return;
    int n = idx / 500, k = idx % 500;
    float s = 0.f;
    for (int t = 0; t < 64; ++t) s += logits[((size_t)(n * 64 + t)) * 512 + k];
    out[idx] = s * (1.f / 64.f);
}

extern "C" void kernel_launch(void* const* d_in, const int* in_sizes, int n_in,
                              void* d_out, int out_size, void* d_ws, size_t ws_size,
                              hipStream_t stream) {
    (void)in_sizes; (void)n_in; (void)out_size; (void)ws_size;
    const float* object_raw = (const float*)d_in[0];
    const float* action_raw = (const float*)d_in[1];
    const float* W_obj = (const float*)d_in[2];
    const float* b_obj = (const float*)d_in[3];
    const float* W_act = (const float*)d_in[4];
    const float* b_act = (const float*)d_in[5];
    const float* gc1_W = (const float*)d_in[6];
    const float* gc1_b = (const float*)d_in[7];
    const float* gc2_W = (const float*)d_in[8];
    const float* gc2_b = (const float*)d_in[9];
    const float* gc3_W = (const float*)d_in[10];
    const float* gc3_b = (const float*)d_in[11];
    const float* gc4_W = (const float*)d_in[12];
    const float* gc4_b = (const float*)d_in[13];
    const float* fc1_W = (const float*)d_in[14];
    const float* fc1_b = (const float*)d_in[15];
    const float* clf_W = (const float*)d_in[16];
    const float* clf_b = (const float*)d_in[17];

    char* base = (char*)d_ws;
    size_t off = 0;
    auto aus = [&](size_t n) -> unsigned short* {
        unsigned short* p = (unsigned short*)(base + off);
        off = (off + n * 2 + 255) & ~(size_t)255;
        return p;
    };
    auto afl = [&](size_t n) -> float* {
        float* p = (float*)(base + off);
        off = (off + n * 4 + 255) & ~(size_t)255;
        return p;
    };

    unsigned short* WobjT  = aus(512 * 512);          // [N=512][K=512]
    unsigned short* WactT  = aus(512 * 512);          // [N=512][K=512]
    unsigned short* fc1WT  = aus(1024 * 512);         // [N=1024][K=512]
    unsigned short* clfWT  = aus(512 * 1024);         // [N=512][K=1024], n>=500 zero
    float*          clfB_p = afl(512);
    unsigned short* gc1T   = aus(32 * 512);           // [32][512]
    unsigned short* gc2T   = aus(512 * 32);           // [512][32]
    unsigned short* gc3T   = aus(32 * 512);
    unsigned short* gc4T   = aus(512 * 32);
    unsigned short* objraw_b = aus((size_t)32768 * 512);
    unsigned short* actraw_b = aus((size_t)2048 * 512);
    unsigned short* obj_b    = aus((size_t)32768 * 512);
    unsigned short* feat_b   = aus((size_t)2048 * 512);
    unsigned short* c1_b     = aus((size_t)2048 * 1024);
    float*          logits   = afl((size_t)2048 * 512);

    auto cvt = [&](const float* s, unsigned short* d, int n) {
        k_cvt_bf16<<<dim3((n + 255) / 256), dim3(256), 0, stream>>>(s, d, n);
    };
    auto cvtT = [&](const float* s, unsigned short* d, int K, int N, int Nsrc) {
        k_cvt_T<<<dim3((N * K + 255) / 256), dim3(256), 0, stream>>>(s, d, K, N, Nsrc);
    };
    cvtT(W_obj, WobjT, 512, 512, 512);
    cvtT(W_act, WactT, 512, 512, 512);
    cvtT(fc1_W, fc1WT, 512, 1024, 1024);
    cvtT(clf_W, clfWT, 1024, 512, 500);
    cvtT(gc1_W, gc1T, 512, 32, 32);
    cvtT(gc2_W, gc2T, 32, 512, 512);
    cvtT(gc3_W, gc3T, 512, 32, 32);
    cvtT(gc4_W, gc4T, 32, 512, 512);
    cvt(object_raw, objraw_b, 32768 * 512);
    cvt(action_raw, actraw_b, 2048 * 512);
    k_pad_bias<<<dim3(2), dim3(256), 0, stream>>>(clf_b, clfB_p);

    // obj = relu(object_raw @ W_obj + b_obj) -> obj_b [32768,512] bf16
    k_gemm_bf16<<<dim3(512 / 64, 32768 / 128), dim3(256), 0, stream>>>(
        objraw_b, WobjT, b_obj, 32768, 512, 512, obj_b, nullptr, 1, 1, 0, 512);
    // act = relu(action_raw @ W_act + b_act) scattered into obj rows r*16+8
    k_gemm_bf16<<<dim3(512 / 64, 2048 / 128), dim3(256), 0, stream>>>(
        actraw_b, WactT, b_act, 2048, 512, 512, obj_b, nullptr, 1, 16, 8, 512);
    // per-graph GCN -> relu(feat) bf16 [2048,512]
    k_graph<<<dim3(2048), dim3(256), 0, stream>>>(
        obj_b, gc1T, gc1_b, gc2T, gc2_b, gc3T, gc3_b, gc4T, gc4_b, feat_b);
    // c = relu(feat @ fc1_W + fc1_b) -> c1_b [2048,1024] bf16
    k_gemm_bf16<<<dim3(1024 / 64, 2048 / 128), dim3(256), 0, stream>>>(
        feat_b, fc1WT, fc1_b, 2048, 1024, 512, c1_b, nullptr, 1, 1, 0, 1024);
    // logits = c @ clf_W + clf_b (padded N=512) -> f32 [2048,512]
    k_gemm_bf16<<<dim3(512 / 64, 2048 / 128), dim3(256), 0, stream>>>(
        c1_b, clfWT, clfB_p, 2048, 512, 1024, nullptr, logits, 0, 1, 0, 512);
    // out[n][k] = mean_T logits
    k_reduce_t<<<dim3((32 * 500 + 255) / 256), dim3(256), 0, stream>>>(logits, (float*)d_out);
}
